// EncoderBlock_13039520711360
// MI455X (gfx1250) — matrix operand order repaired and checked by reading, not compiled, not run on hardware
//
#include <hip/hip_runtime.h>
#include <hip/hip_bf16.h>

// ---------------------------------------------------------------------------
// CDNA5 (gfx1250) encoder block: LN -> QKV (WMMA) -> flash-attn (WMMA) ->
// O-proj (+residual) -> LN -> FFN (WMMA, ReLU) -> +residual.
// GEMMs use v_wmma_f32_16x16x32_bf16 (fp32 -> bf16 inputs, f32 accum).
// A-tile staging in the GEMM uses the Tensor Data Mover (tensor_load_to_lds,
// TENSORcnt).  This toolchain exposes the 6-arg builtin form:
//   (u32x4 g0, i32x8 g1, i32x4 g2, i32x4 g3, i32x8 pad, i32 cpol)
// B=2, S=2048, D=1024, H=16, dk=64, F=4096.
// ---------------------------------------------------------------------------

typedef __bf16 bf16;
typedef __bf16 v16bf __attribute__((ext_vector_type(16)));
typedef float  v8f   __attribute__((ext_vector_type(8)));
typedef unsigned int u32x4 __attribute__((ext_vector_type(4)));
typedef int          i32x8 __attribute__((ext_vector_type(8)));
typedef int          i32x4 __attribute__((ext_vector_type(4)));

#if defined(__has_builtin)
#if __has_builtin(__builtin_amdgcn_tensor_load_to_lds) && \
    __has_builtin(__builtin_amdgcn_s_wait_tensorcnt)
#define HAVE_TDM 1
#endif
#endif
#ifndef HAVE_TDM
#define HAVE_TDM 0
#endif

__device__ __forceinline__ v8f wmma_bf16(v16bf a, v16bf b, v8f c) {
  // (neg_a, A, neg_b, B, c_mod, C, reuse_a, reuse_b)
  return __builtin_amdgcn_wmma_f32_16x16x32_bf16(false, a, false, b, (short)0, c,
                                                 false, false);
}

// A-matrix fragment (16x32 bf16, MxK): lane L holds row (L&15);
// K values: kb..kb+15 contiguous where kb = (L>>4)*16.
__device__ __forceinline__ v16bf load_a_frag(const bf16* base, int ld, int lane) {
  const int row = lane & 15;
  const int kb  = (lane >> 4) * 16;
  const bf16* p = base + row * ld + kb;
  v16bf a;
#pragma unroll
  for (int j = 0; j < 8; ++j) a[j] = p[j];
#pragma unroll
  for (int j = 0; j < 8; ++j) a[8 + j] = p[8 + j];
  return a;
}

// B-matrix fragment (32x16 bf16, KxN) sourced from an [N][K] row-major array:
// lane L holds column n = (L&15); K values kb2..kb2+15 contiguous,
// kb2 = (L>>4)*16.  B[k][n] == W[n][k], so a weight row supplies the lane.
__device__ __forceinline__ v16bf load_b_frag(const bf16* base, int ld, int lane) {
  const int n   = lane & 15;
  const int kb2 = (lane >> 4) * 16;
  const bf16* p = base + n * ld + kb2;
  v16bf b;
#pragma unroll
  for (int j = 0; j < 16; ++j) b[j] = p[j];
  return b;
}

#if HAVE_TDM
// TDM 2D tile load: rows x cols (bf16) from global (row stride in elements)
// into LDS at lds_off, inserting 4 DWORDs of pad after every 16 DWORDs
// (i.e. 32 bf16 row -> 40 bf16 LDS pitch).  cols must be 32 here.
__device__ __forceinline__ void tdm_load_tile_bf16(const bf16* gsrc,
                                                   unsigned lds_off, int rows,
                                                   int cols, long row_stride) {
  const unsigned long long ga = (unsigned long long)(size_t)gsrc;
  const unsigned long long st = (unsigned long long)row_stride;
  u32x4 g0;
  g0[0] = 1u;                                   // count=1 (valid user D#)
  g0[1] = lds_off;                              // lds_addr (bytes)
  g0[2] = (unsigned)(ga & 0xffffffffull);       // global_addr[31:0]
  g0[3] = (unsigned)((ga >> 32) & 0x01ffffffull) | (2u << 30);  // [56:32]|type=2
  i32x8 g1;
  // data_size=1 (2B), pad_enable, pad_interval=3 (16 dw), pad_amount=3 (4 dw)
  g1[0] = (int)((1u << 16) | (1u << 20) | (3u << 22) | (3u << 25));
  g1[1] = (int)(((unsigned)cols & 0xffffu) << 16);              // tensor_dim0 lo
  g1[2] = (int)((((unsigned)cols >> 16) & 0xffffu) |
                (((unsigned)rows & 0xffffu) << 16));            // dim0 hi|dim1 lo
  g1[3] = (int)((((unsigned)rows >> 16) & 0xffffu) |
                (((unsigned)cols & 0xffffu) << 16));            // dim1 hi|tile0
  g1[4] = (int)((unsigned)rows & 0xffffu);                      // tile1 (tile2=0)
  g1[5] = (int)(st & 0xffffffffull);                            // dim0_stride lo
  g1[6] = (int)((st >> 32) & 0xffffull);                        // stride hi
  g1[7] = 0;
  const i32x4 z4 = {0, 0, 0, 0};
  const i32x8 z8 = {0, 0, 0, 0, 0, 0, 0, 0};
  // 6-arg form (this toolchain). ROCm 7.2 / clang-22 would be the 5-arg form:
  //   __builtin_amdgcn_tensor_load_to_lds(g0, g1, z4, z4, 0);
  __builtin_amdgcn_tensor_load_to_lds(g0, g1, z4, z4, z8, 0);
}
#endif

// ---------------------------------------------------------------------------
// LayerNorm (mean over D, std with ddof=1, scalar alpha/beta) -> bf16 out.
// ---------------------------------------------------------------------------
__global__ __launch_bounds__(256) void ln_kernel(const float* __restrict__ x,
                                                 const float* __restrict__ alpha,
                                                 const float* __restrict__ beta,
                                                 bf16* __restrict__ out, int D) {
  const int row = blockIdx.x;
  const int tid = threadIdx.x;
  const float* xr = x + (size_t)row * D;
  float s = 0.f, s2 = 0.f;
  for (int i = tid; i < D; i += 256) {
    float v = xr[i];
    s += v;
    s2 += v * v;
  }
  __shared__ float sb[256];
  __shared__ float qb[256];
  sb[tid] = s;
  qb[tid] = s2;
  __syncthreads();
  for (int off = 128; off > 0; off >>= 1) {
    if (tid < off) {
      sb[tid] += sb[tid + off];
      qb[tid] += qb[tid + off];
    }
    __syncthreads();
  }
  const float mean = sb[0] / (float)D;
  const float var  = (qb[0] - (float)D * mean * mean) / (float)(D - 1);
  const float inv  = alpha[0] / (sqrtf(var) + 1e-6f);
  const float bo   = beta[0];
  bf16* orow = out + (size_t)row * D;
  for (int i = tid; i < D; i += 256) orow[i] = (bf16)((xr[i] - mean) * inv + bo);
}

// ---------------------------------------------------------------------------
// GEMM: C[M,N] = act(A[M,K] (bf16) @ W[N,K]^T (f32->bf16) + bias) [+ res]
// Block tile 128x64, BK=32. 8 waves, each 32x32 (2x2 WMMA tiles).
// A tile staged by TDM (wave 0 issues tensor_load_to_lds), W staged manually
// (needs f32->bf16 conversion).
// ---------------------------------------------------------------------------
template <bool RELU, bool RES, bool F32OUT, bool BF16OUT>
__global__ __launch_bounds__(256) void gemm_kernel(
    const bf16* __restrict__ A, const float* __restrict__ W,
    const float* __restrict__ bias, const float* __restrict__ res,
    float* __restrict__ outf, bf16* __restrict__ outb, int M, int N, int K) {
  constexpr int BM = 128, BN = 64, BK = 32, LDA = BK + 8, LDB = BK + 8;
  __shared__ bf16 As[BM][LDA];
  __shared__ bf16 Bs[BN][LDB];

  const int bm   = blockIdx.y * BM;
  const int bn   = blockIdx.x * BN;
  const int tid  = threadIdx.x;
  const int lane = tid & 31;
  const int wid  = tid >> 5;
  const int wm   = (wid >> 1) * 32;  // 0,32,64,96
  const int wn   = (wid & 1) * 32;   // 0,32

  v8f acc[2][2];
#pragma unroll
  for (int i = 0; i < 2; ++i)
#pragma unroll
    for (int j = 0; j < 2; ++j)
      acc[i][j] = (v8f){0.f, 0.f, 0.f, 0.f, 0.f, 0.f, 0.f, 0.f};

#if HAVE_TDM
  const unsigned as_lds = (unsigned)(size_t)&As[0][0];
#endif

  for (int k0 = 0; k0 < K; k0 += BK) {
    __syncthreads();
#if HAVE_TDM
    if (wid == 0) {  // wave-uniform: one TDM transfer for the whole A tile
      tdm_load_tile_bf16(A + (size_t)bm * K + k0, as_lds, BM, BK, (long)K);
    }
#else
    {  // stage A tile: 128 rows x 32 k (bf16), 16 elems/thread
      const int r = tid >> 1, c = (tid & 1) * 16;
      const bf16* g = A + (size_t)(bm + r) * K + k0 + c;
#pragma unroll
      for (int j = 0; j < 16; ++j) As[r][c + j] = g[j];
    }
#endif
    {  // stage W tile: 64 rows x 32 k (f32 -> bf16), 8 elems/thread
      const int r = tid >> 2, c = (tid & 3) * 8;
      const float* g = W + (size_t)(bn + r) * K + k0 + c;
#pragma unroll
      for (int j = 0; j < 8; ++j) Bs[r][c + j] = (bf16)g[j];
      if (k0 + BK < K) __builtin_prefetch(g + BK, 0, 1);  // global_prefetch_b8
    }
#if HAVE_TDM
    if (wid == 0) __builtin_amdgcn_s_wait_tensorcnt(0);
#endif
    __syncthreads();
    v16bf bfr[2];
#pragma unroll
    for (int nt = 0; nt < 2; ++nt)
      bfr[nt] = load_b_frag(&Bs[wn + nt * 16][0], LDB, lane);
#pragma unroll
    for (int mt = 0; mt < 2; ++mt) {
      v16bf afr = load_a_frag(&As[wm + mt * 16][0], LDA, lane);
#pragma unroll
      for (int nt = 0; nt < 2; ++nt)
        acc[mt][nt] = wmma_bf16(afr, bfr[nt], acc[mt][nt]);
    }
  }

  // Epilogue: C/D layout -> VGPR r holds row (r + 8*(lane>>4)), col (lane&15).
  const int rbase = bm + wm + (lane >> 4) * 8;
  const int cbase = bn + wn + (lane & 15);
#pragma unroll
  for (int mt = 0; mt < 2; ++mt)
#pragma unroll
    for (int nt = 0; nt < 2; ++nt)
#pragma unroll
      for (int r = 0; r < 8; ++r) {
        const int row = rbase + mt * 16 + r;
        const int col = cbase + nt * 16;
        float v = acc[mt][nt][r] + bias[col];
        if (RELU) v = fmaxf(v, 0.f);
        if (RES) v += res[(size_t)row * N + col];
        if (F32OUT) outf[(size_t)row * N + col] = v;
        if (BF16OUT) outb[(size_t)row * N + col] = (bf16)v;
      }
}

// ---------------------------------------------------------------------------
// Flash attention (no 1/sqrt(dk) scale, matching reference).
// Grid (S/128, H, B); 8 waves per block, each wave owns 16 query rows.
// ---------------------------------------------------------------------------
__global__ __launch_bounds__(256) void attn_kernel(
    const bf16* __restrict__ Q, const bf16* __restrict__ Km,
    const bf16* __restrict__ V, const int* __restrict__ mask,
    bf16* __restrict__ O, int S, int D) {
  constexpr int DK = 64;
  const int qc   = blockIdx.x * 128;
  const int h    = blockIdx.y;
  const int b    = blockIdx.z;
  const int tid  = threadIdx.x;
  const int lane = tid & 31;
  const int wid  = tid >> 5;
  const int q0   = qc + wid * 16;

  __shared__ bf16 Ks[32][DK + 8];    // [key][d]
  __shared__ bf16 Vt[DK][32 + 8];    // [d][key] (transposed on stage)
  __shared__ bf16 P[8][16][32 + 8];  // per-wave P tile (q x key)

  const bf16* qbase = Q + ((size_t)b * S + q0) * D + h * DK;
  v16bf qa[2];
  qa[0] = load_a_frag(qbase, D, lane);       // d = 0..31
  qa[1] = load_a_frag(qbase + 32, D, lane);  // d = 32..63

  float mrun[8], lrun[8];
  v8f oacc[4];
#pragma unroll
  for (int r = 0; r < 8; ++r) { mrun[r] = -3.0e38f; lrun[r] = 0.f; }
#pragma unroll
  for (int dt = 0; dt < 4; ++dt)
    oacc[dt] = (v8f){0.f, 0.f, 0.f, 0.f, 0.f, 0.f, 0.f, 0.f};

  const int* mrow = mask + (size_t)b * S;

  for (int kc = 0; kc < S; kc += 32) {
    __syncthreads();
    {  // stage K chunk (32x64) and transposed V chunk
      const int r = tid >> 3, c = (tid & 7) * 8;
      const bf16* gk = Km + ((size_t)b * S + kc + r) * D + h * DK + c;
      const bf16* gv = V + ((size_t)b * S + kc + r) * D + h * DK + c;
#pragma unroll
      for (int j = 0; j < 8; ++j) Ks[r][c + j] = gk[j];
#pragma unroll
      for (int j = 0; j < 8; ++j) Vt[c + j][r] = gv[j];
    }
    __syncthreads();

    // --- scores: 16q x 32k via 4 WMMAs ---
    v8f sc[2];
#pragma unroll
    for (int nt = 0; nt < 2; ++nt) {
      v8f s = (v8f){0.f, 0.f, 0.f, 0.f, 0.f, 0.f, 0.f, 0.f};
#pragma unroll
      for (int ks = 0; ks < 2; ++ks) {
        v16bf bfr = load_b_frag(&Ks[nt * 16][ks * 32], DK + 8, lane);
        s = wmma_bf16(qa[ks], bfr, s);
      }
      const int key = kc + nt * 16 + (lane & 15);
      if (mrow[key] == 0) {
#pragma unroll
        for (int r = 0; r < 8; ++r) s[r] = -3.0e38f;
      }
      sc[nt] = s;
    }

    // --- online softmax (rows in one 16-lane half; cols across lanes) ---
    float scale[8];
#pragma unroll
    for (int r = 0; r < 8; ++r) {
      float t = fmaxf(sc[0][r], sc[1][r]);
#pragma unroll
      for (int m = 8; m >= 1; m >>= 1) t = fmaxf(t, __shfl_xor(t, m, 16));
      const float mnew = fmaxf(mrun[r], t);
      const float fac  = __expf(mrun[r] - mnew);
      const float p0   = __expf(sc[0][r] - mnew);
      const float p1   = __expf(sc[1][r] - mnew);
      float ps = p0 + p1;
#pragma unroll
      for (int m = 8; m >= 1; m >>= 1) ps += __shfl_xor(ps, m, 16);
      lrun[r]  = lrun[r] * fac + ps;
      mrun[r]  = mnew;
      scale[r] = fac;
      sc[0][r] = p0;
      sc[1][r] = p1;
    }
#pragma unroll
    for (int dt = 0; dt < 4; ++dt)
#pragma unroll
      for (int r = 0; r < 8; ++r) oacc[dt][r] *= scale[r];

    // --- spill P (bf16) to wave-private LDS for use as WMMA A operand ---
    {
      const int pr = (lane >> 4) * 8;
      const int pc = lane & 15;
#pragma unroll
      for (int r = 0; r < 8; ++r) {
        P[wid][pr + r][pc]      = (bf16)sc[0][r];
        P[wid][pr + r][16 + pc] = (bf16)sc[1][r];
      }
    }

    // --- O += P @ V (4 WMMAs; V fragment from transposed LDS chunk) ---
    v16bf pa = load_a_frag(&P[wid][0][0], 32 + 8, lane);
#pragma unroll
    for (int dt = 0; dt < 4; ++dt) {
      v16bf vb = load_b_frag(&Vt[dt * 16][0], 32 + 8, lane);
      oacc[dt] = wmma_bf16(pa, vb, oacc[dt]);
    }
  }

  // epilogue: normalize and store bf16 [B*S, D]
#pragma unroll
  for (int dt = 0; dt < 4; ++dt)
#pragma unroll
    for (int r = 0; r < 8; ++r) {
      const int row = q0 + (lane >> 4) * 8 + r;
      const float v = oacc[dt][r] / lrun[r];
      O[((size_t)b * S + row) * D + h * DK + dt * 16 + (lane & 15)] = (bf16)v;
    }
}

// ---------------------------------------------------------------------------
extern "C" void kernel_launch(void* const* d_in, const int* in_sizes, int n_in,
                              void* d_out, int out_size, void* d_ws, size_t ws_size,
                              hipStream_t stream) {
  (void)in_sizes; (void)n_in; (void)out_size; (void)ws_size;
  constexpr int B = 2, S = 2048, D = 1024, F = 4096, H = 16;
  constexpr int M = B * S;

  const float* x      = (const float*)d_in[0];
  const int*   mask   = (const int*)d_in[1];
  const float* wq     = (const float*)d_in[2];
  const float* bq     = (const float*)d_in[3];
  const float* wk     = (const float*)d_in[4];
  const float* bk     = (const float*)d_in[5];
  const float* wv     = (const float*)d_in[6];
  const float* bv     = (const float*)d_in[7];
  const float* wo     = (const float*)d_in[8];
  const float* bo     = (const float*)d_in[9];
  const float* w1     = (const float*)d_in[10];
  const float* b1     = (const float*)d_in[11];
  const float* w2     = (const float*)d_in[12];
  const float* b2     = (const float*)d_in[13];
  const float* alpha1 = (const float*)d_in[14];
  const float* beta1  = (const float*)d_in[15];
  const float* alpha2 = (const float*)d_in[16];
  const float* beta2  = (const float*)d_in[17];
  float* out = (float*)d_out;

  // Workspace layout (bytes):
  //   [0,   8M)  xn   bf16 [M,D]   (ln1 then ln2 output)
  //   [8M, 16M)  q    bf16 [M,D]
  //   [16M,24M)  k    bf16 [M,D]
  //   [24M,32M)  v    bf16 [M,D]
  //   [32M,40M)  attn bf16 [M,D]
  //   [40M,56M)  x2   f32  [M,D]
  //   h bf16 [M,F] (32MB) reuses [8M,40M) once q/k/v/attn are dead.
  char* ws = (char*)d_ws;
  const size_t MB = 1024ull * 1024ull;
  bf16*  xn   = (bf16*)(ws + 0 * MB);
  bf16*  q    = (bf16*)(ws + 8 * MB);
  bf16*  k    = (bf16*)(ws + 16 * MB);
  bf16*  v    = (bf16*)(ws + 24 * MB);
  bf16*  attn = (bf16*)(ws + 32 * MB);
  float* x2   = (float*)(ws + 40 * MB);
  bf16*  hbuf = (bf16*)(ws + 8 * MB);

  const dim3 blk(256);
  const dim3 gemmD(D / 64, M / 128);  // N=1024
  const dim3 gemmF(F / 64, M / 128);  // N=4096
  const dim3 attng(S / 128, H, B);

  ln_kernel<<<M, blk, 0, stream>>>(x, alpha1, beta1, xn, D);
  gemm_kernel<false, false, false, true><<<gemmD, blk, 0, stream>>>(
      xn, wq, bq, nullptr, nullptr, q, M, D, D);
  gemm_kernel<false, false, false, true><<<gemmD, blk, 0, stream>>>(
      xn, wk, bk, nullptr, nullptr, k, M, D, D);
  gemm_kernel<false, false, false, true><<<gemmD, blk, 0, stream>>>(
      xn, wv, bv, nullptr, nullptr, v, M, D, D);
  attn_kernel<<<attng, blk, 0, stream>>>(q, k, v, mask, attn, S, D);
  gemm_kernel<false, true, true, false><<<gemmD, blk, 0, stream>>>(
      attn, wo, bo, x, x2, nullptr, M, D, D);
  ln_kernel<<<M, blk, 0, stream>>>(x2, alpha2, beta2, xn, D);
  gemm_kernel<true, false, false, true><<<gemmF, blk, 0, stream>>>(
      xn, w1, b1, nullptr, nullptr, hbuf, M, F, D);
  gemm_kernel<false, true, true, false><<<gemmD, blk, 0, stream>>>(
      hbuf, w2, b2, x2, out, nullptr, M, D, F);
}
